// SparseAttention_35656818492142
// MI455X (gfx1250) — matrix operand order repaired and checked by reading, not compiled, not run on hardware
//
#include <hip/hip_runtime.h>

// ---------------------------------------------------------------------------
// SparseAttention for MI455X (gfx1250, wave32, WMMA).
// bf16 WMMA (v_wmma_f32_16x16x32_bf16) for all three matmul stages, f32
// accumulation, flash-style online softmax. Cross-lane row reductions use
// v_permlane16_b32 (VALU) instead of ds_bpermute to keep the LDS pipe free
// for the P-transpose / V^T staging. global_prefetch_b8 hides next-tile
// latency in both the GEMM K-loop and the attention chunk loop.
// Workspace ~122 MB (see kernel_launch).
// ---------------------------------------------------------------------------

#define B_   128
#define T_   256
#define C_   384
#define M_   512
#define H_   8
#define HD_  48
#define WIN_ 64
#define SCALE_ 0.144337567297406f   // 48^-0.5

typedef __attribute__((ext_vector_type(16))) __bf16         v16bf;
typedef __attribute__((ext_vector_type(16))) unsigned short v16u;
typedef __attribute__((ext_vector_type(8)))  float          v8f;
typedef __attribute__((ext_vector_type(4)))  unsigned int   u32x4;

union AV {                       // one WMMA 16-bit operand (8 VGPRs / lane)
  v16u           u;
  u32x4          q[2];
  unsigned short s[16];
};

__device__ __forceinline__ unsigned short f2bf(float f) {
  unsigned u = __builtin_bit_cast(unsigned, f);
  u += 0x7fffu + ((u >> 16) & 1u);           // round-to-nearest-even
  return (unsigned short)(u >> 16);
}

__device__ __forceinline__ v8f wmma_bf16(const AV& a, const AV& b, v8f c) {
  // D(f32 16x16) = A(bf16 16x32) * B(bf16 32x16) + C
  return __builtin_amdgcn_wmma_f32_16x16x32_bf16(
      false, __builtin_bit_cast(v16bf, a.u),
      false, __builtin_bit_cast(v16bf, b.u),
      (short)0, c, false, false);
}

__device__ __forceinline__ void lds_wait() {
#if __has_builtin(__builtin_amdgcn_s_wait_dscnt)
  __builtin_amdgcn_s_wait_dscnt(0);
#else
  asm volatile("s_wait_dscnt 0x0" ::: "memory");
#endif
}

// --- xor-by-K lane shuffle within 16-lane groups, on the VALU pipe ---------
constexpr unsigned selpack(int base, int K) {
  unsigned r = 0;
  for (int i = 0; i < 8; i++) r |= ((unsigned)(((base + i) ^ K) & 0xF)) << (4 * i);
  return r;
}

template <int K>
__device__ __forceinline__ float permxor16(float x) {
#if __has_builtin(__builtin_amdgcn_permlane16)
  constexpr unsigned s0 = selpack(0, K);
  constexpr unsigned s1 = selpack(8, K);
  unsigned u = __builtin_bit_cast(unsigned, x);
  u = __builtin_amdgcn_permlane16(u, u, s0, s1, false, false);
  return __builtin_bit_cast(float, u);
#else
  return __shfl_xor(x, K);
#endif
}

__device__ __forceinline__ float rowmax16(float x) {
  x = fmaxf(x, permxor16<1>(x));
  x = fmaxf(x, permxor16<2>(x));
  x = fmaxf(x, permxor16<4>(x));
  x = fmaxf(x, permxor16<8>(x));
  return x;
}
__device__ __forceinline__ float rowsum16(float x) {
  x += permxor16<1>(x);
  x += permxor16<2>(x);
  x += permxor16<4>(x);
  x += permxor16<8>(x);
  return x;
}

// ---------------------------------------------------------------------------
// f32 -> bf16 conversion
// ---------------------------------------------------------------------------
__global__ void cvt_bf16_kernel(const float* __restrict__ src,
                                unsigned short* __restrict__ dst, int n) {
  for (int i = blockIdx.x * blockDim.x + threadIdx.x; i < n;
       i += gridDim.x * blockDim.x)
    dst[i] = f2bf(src[i]);
}

// ---------------------------------------------------------------------------
// C[M,N] = A[M,K](bf16) @ B[K,N](bf16) + bias[N]
//   OUT_BF16=1 : store bf16
//   OUT_BF16=0 : store f32, scaled by rowscale[m] (loss mask)
// 256 threads = 8 waves; macro-tile 128(M) x 64(N); K stepped by 32.
// ---------------------------------------------------------------------------
template <int OUT_BF16>
__global__ __launch_bounds__(256)
void gemm_bf16_kernel(const unsigned short* __restrict__ A,
                      const unsigned short* __restrict__ Bw,
                      const float* __restrict__ bias,
                      const float* __restrict__ rowscale,
                      void* __restrict__ Cout, int M, int N, int K) {
  __shared__ __align__(16) unsigned short sA[128 * 32];   // A tile, row-major
  __shared__ __align__(16) unsigned short sBt[64 * 32];   // B tile, transposed

  const int tid  = threadIdx.x;
  const int lane = tid & 31;
  const int wv   = tid >> 5;
  const int l15  = lane & 15;
  const int hi   = lane >> 4;
  const int ko   = hi * 8;
  const int mb   = blockIdx.y * 128;
  const int nb   = blockIdx.x * 64;

  const int rA = tid >> 1, cA = (tid & 1) * 16;
  const int kB = tid >> 3, nB = (tid & 7) * 8;

  v8f acc[4] = {};

  for (int k0 = 0; k0 < K; k0 += 32) {
    {  // stage A 128x32 (two 16B vectors per thread)
      const u32x4* g = (const u32x4*)(A + (size_t)(mb + rA) * K + k0 + cA);
      *(u32x4*)&sA[rA * 32 + cA]     = g[0];
      *(u32x4*)&sA[rA * 32 + cA + 8] = g[1];
    }
    {  // stage B 32x64 transposed -> sBt[n][k]
      u32x4 d = *(const u32x4*)(Bw + (size_t)(k0 + kB) * N + nb + nB);
      unsigned short e[8];
      *(u32x4*)e = d;
#pragma unroll
      for (int q = 0; q < 8; q++) sBt[(nB + q) * 32 + kB] = e[q];
    }
    if (k0 + 32 < K) {  // prefetch next K-step tiles into WGP$/L2
      __builtin_prefetch(A + (size_t)(mb + rA) * K + k0 + 32 + cA, 0, 3);
      __builtin_prefetch(Bw + (size_t)(k0 + 32 + kB) * N + nb + nB, 0, 3);
    }
    __syncthreads();

    AV a;  // A tile rows 16*wv .. 16*wv+15
    {
      int arow = wv * 16 + l15;
      a.q[0] = *(const u32x4*)&sA[arow * 32 + ko];
      a.q[1] = *(const u32x4*)&sA[arow * 32 + 16 + ko];
    }
#pragma unroll
    for (int nt = 0; nt < 4; nt++) {
      AV b;
      int n = nt * 16 + l15;
      b.q[0]  = *(const u32x4*)&sBt[n * 32 + ko];
      b.q[1]  = *(const u32x4*)&sBt[n * 32 + 16 + ko];
      acc[nt] = wmma_bf16(a, b, acc[nt]);
    }
    __syncthreads();
  }

  // epilogue: D element v -> (row = v + 8*hi, col = l15)
#pragma unroll
  for (int nt = 0; nt < 4; nt++) {
    int n    = nb + nt * 16 + l15;
    float bz = bias[n];
#pragma unroll
    for (int v = 0; v < 8; v++) {
      int m     = mb + wv * 16 + hi * 8 + v;
      float val = acc[nt][v] + bz;
      if (OUT_BF16)
        ((unsigned short*)Cout)[(size_t)m * N + n] = f2bf(val);
      else
        ((float*)Cout)[(size_t)m * N + n] = val * rowscale[m];
    }
  }
}

// ---------------------------------------------------------------------------
// Attention: one workgroup per (b,h); 4 waves, each owns q-tiles qt, qt+4, ...
// Per 32-key chunk: S = QK^T (4 wmma, hd padded 48->64), online softmax,
// P staged via LDS (D-layout -> A-layout transpose), V staged transposed in
// LDS, PV (3 wmma). Gate fused into bf16 output.
// ---------------------------------------------------------------------------
__device__ __forceinline__ void attn_chunk(
    const unsigned short* __restrict__ kbase,  // row of first key in chunk
    const unsigned short* __restrict__ vbase,
    int rstride, int selfj0 /* -1 = mem (unmasked) */, int i0,
    const AV& a0, const AV& a1, float* mrow, float* lrow, v8f* acc,
    unsigned short* P, unsigned short* Vt, int lane) {
  const int l15 = lane & 15, hi = lane >> 4, ko = hi * 8;

  // ---- stage V^T into LDS: Vt[d][key], lane owns one key row -------------
  {
    int voff = lane;
    if (selfj0 >= 0 && selfj0 + voff > T_ - 1) voff = T_ - 1 - selfj0;
    const unsigned short* vr = vbase + (size_t)voff * rstride;
#pragma unroll
    for (int c8 = 0; c8 < 6; c8++) {
      unsigned short e[8];
      *(u32x4*)e = *(const u32x4*)(vr + c8 * 8);
#pragma unroll
      for (int q = 0; q < 8; q++) Vt[(c8 * 8 + q) * 32 + lane] = e[q];
    }
  }

  // ---- S = Q K^T for the two 16-key halves -------------------------------
  v8f s[2];
#pragma unroll
  for (int st = 0; st < 2; st++) {
    int roff = st * 16 + l15;
    if (selfj0 >= 0 && selfj0 + roff > T_ - 1) roff = T_ - 1 - selfj0;
    const unsigned short* kr = kbase + (size_t)roff * rstride;
    AV b0, b1;
    b0.q[0] = *(const u32x4*)(kr + ko);               // d = 8*hi .. +7, 16+8*hi .. +7
    b0.q[1] = *(const u32x4*)(kr + 16 + ko);
    {                                                 // d = 32+8*hi .. +7 (rest pad)
      u32x4 z = {0u, 0u, 0u, 0u};
      b1.q[0] = *(const u32x4*)(kr + 32 + ko);
      b1.q[1] = z;
    }
    v8f z = {};
    z     = wmma_bf16(a0, b0, z);
    z     = wmma_bf16(a1, b1, z);
    s[st] = z;
  }

  // ---- scale + mask + chunk row-max (permlane16 butterflies) -------------
  float cmax[8];
#pragma unroll
  for (int v = 0; v < 8; v++) {
    float s0 = s[0][v] * SCALE_, s1 = s[1][v] * SCALE_;
    if (selfj0 >= 0) {
      int i  = i0 + v + 8 * hi;
      int j0 = selfj0 + l15, j1 = selfj0 + 16 + l15;
      if (!(j0 <= i && i - j0 <= WIN_)) s0 = -1e30f;
      if (!(j1 <= i && i - j1 <= WIN_)) s1 = -1e30f;
    }
    s[0][v] = s0;
    s[1][v] = s1;
    cmax[v] = rowmax16(fmaxf(s0, s1));
  }

  // ---- online softmax update ---------------------------------------------
#pragma unroll
  for (int v = 0; v < 8; v++) {
    float nm   = fmaxf(mrow[v], cmax[v]);
    float corr = __expf(mrow[v] - nm);
    mrow[v]    = nm;
    float p0 = __expf(s[0][v] - nm);
    float p1 = __expf(s[1][v] - nm);
    s[0][v] = p0;
    s[1][v] = p1;
    lrow[v] = lrow[v] * corr + rowsum16(p0 + p1);
    acc[0][v] *= corr;
    acc[1][v] *= corr;
    acc[2][v] *= corr;
  }

  // ---- P : D-layout -> A-layout via LDS ----------------------------------
#pragma unroll
  for (int st = 0; st < 2; st++)
#pragma unroll
    for (int v = 0; v < 8; v++)
      P[(v + 8 * hi) * 32 + st * 16 + l15] = f2bf(s[st][v]);
  lds_wait();

  AV pa;
  pa.q[0] = *(const u32x4*)&P[l15 * 32 + ko];
  pa.q[1] = *(const u32x4*)&P[l15 * 32 + 16 + ko];

  // ---- O += P V  (3 N-tiles of 16 dims) ----------------------------------
#pragma unroll
  for (int nt = 0; nt < 3; nt++) {
    AV vb;
    int d   = nt * 16 + l15;
    vb.q[0] = *(const u32x4*)&Vt[d * 32 + ko];
    vb.q[1] = *(const u32x4*)&Vt[d * 32 + 16 + ko];
    acc[nt] = wmma_bf16(pa, vb, acc[nt]);
  }
}

__global__ __launch_bounds__(128)
void attn_kernel(const unsigned short* __restrict__ qkv,
                 const unsigned short* __restrict__ memb,
                 const float* __restrict__ gate,
                 unsigned short* __restrict__ y) {
  __shared__ __align__(16) unsigned short sP[4][16 * 32];
  __shared__ __align__(16) unsigned short sVt[4][48 * 32];

  const int lane = threadIdx.x & 31;
  const int wv   = threadIdx.x >> 5;
  const int l15  = lane & 15, hi = lane >> 4, ko = hi * 8;
  const int b = blockIdx.x >> 3, h = blockIdx.x & 7;

  unsigned short* P  = sP[wv];
  unsigned short* Vt = sVt[wv];

  const unsigned short* qb  = qkv + (size_t)b * T_ * 1152 + h * HD_;
  const unsigned short* kb  = qkv + (size_t)b * T_ * 1152 + C_ + h * HD_;
  const unsigned short* vb  = qkv + (size_t)b * T_ * 1152 + 2 * C_ + h * HD_;
  const unsigned short* mhb = memb + h * HD_;

  for (int qt = wv; qt < 16; qt += 4) {
    const int i0 = qt * 16;

    // Q A-operands (hd 0..31 and 32..47 padded to 64)
    const unsigned short* qrow = qb + (size_t)(i0 + l15) * 1152;
    AV a0, a1;
    a0.q[0] = *(const u32x4*)(qrow + ko);
    a0.q[1] = *(const u32x4*)(qrow + 16 + ko);
    a1.q[0] = *(const u32x4*)(qrow + 32 + ko);
    {
      u32x4 z = {0u, 0u, 0u, 0u};
      a1.q[1] = z;
    }

    float mrow[8], lrow[8];
    v8f acc[3] = {};
#pragma unroll
    for (int v = 0; v < 8; v++) { mrow[v] = -1e30f; lrow[v] = 0.f; }

    // memory keys: 512 = 16 chunks of 32, unmasked
    for (int c = 0; c < 16; c++) {
      const unsigned short* kr = mhb + (size_t)(c * 32) * C_;
      if (c + 1 < 16)  // prefetch next chunk's K/V rows (lane-spread)
        __builtin_prefetch(mhb + (size_t)((c + 1) * 32 + lane) * C_, 0, 2);
      attn_chunk(kr, kr, C_, -1, i0, a0, a1, mrow, lrow, acc, P, Vt, lane);
    }
    // windowed causal self keys: only chunks intersecting [i0-64, i0+15]
    int start = (i0 > WIN_) ? ((i0 - WIN_) & ~31) : 0;
    for (int js0 = start; js0 < i0 + 16; js0 += 32) {
      __builtin_prefetch(vb + (size_t)(js0 + lane) * 1152, 0, 2);
      attn_chunk(kb + (size_t)js0 * 1152, vb + (size_t)js0 * 1152, 1152, js0,
                 i0, a0, a1, mrow, lrow, acc, P, Vt, lane);
    }

    // normalize, gate, store bf16
#pragma unroll
    for (int nt = 0; nt < 3; nt++) {
      int c   = h * HD_ + nt * 16 + l15;
      float g = gate[c];
#pragma unroll
      for (int v = 0; v < 8; v++) {
        int i = i0 + v + 8 * hi;
        y[((size_t)b * T_ + i) * C_ + c] = f2bf(acc[nt][v] / lrow[v] * g);
      }
    }
  }
}

// ---------------------------------------------------------------------------
// launch
// ---------------------------------------------------------------------------
extern "C" void kernel_launch(void* const* d_in, const int* in_sizes, int n_in,
                              void* d_out, int out_size, void* d_ws,
                              size_t ws_size, hipStream_t stream) {
  const float* x         = (const float*)d_in[0];
  const float* mem       = (const float*)d_in[1];
  const float* loss_mask = (const float*)d_in[2];
  const float* Wqkv      = (const float*)d_in[3];
  const float* bqkv      = (const float*)d_in[4];
  const float* Wproj     = (const float*)d_in[5];
  const float* bproj     = (const float*)d_in[6];
  const float* gate      = (const float*)d_in[7];
  float* out             = (float*)d_out;

  char* ws                 = (char*)d_ws;
  unsigned short* x_bf     = (unsigned short*)ws; ws += (size_t)B_ * T_ * C_ * 2;
  unsigned short* mem_bf   = (unsigned short*)ws; ws += (size_t)M_ * C_ * 2;
  unsigned short* Wqkv_bf  = (unsigned short*)ws; ws += (size_t)C_ * 3 * C_ * 2;
  unsigned short* Wproj_bf = (unsigned short*)ws; ws += (size_t)C_ * C_ * 2;
  unsigned short* qkv_bf   = (unsigned short*)ws; ws += (size_t)B_ * T_ * 3 * C_ * 2;
  unsigned short* y_bf     = (unsigned short*)ws; ws += (size_t)B_ * T_ * C_ * 2;

  auto cvt = [&](const float* s, unsigned short* d, int n) {
    cvt_bf16_kernel<<<(n + 255) / 256, 256, 0, stream>>>(s, d, n);
  };
  cvt(x, x_bf, B_ * T_ * C_);
  cvt(mem, mem_bf, M_ * C_);
  cvt(Wqkv, Wqkv_bf, C_ * 3 * C_);
  cvt(Wproj, Wproj_bf, C_ * C_);

  // qkv = x @ Wqkv + bqkv  -> bf16
  gemm_bf16_kernel<1><<<dim3((3 * C_) / 64, (B_ * T_) / 128), 256, 0, stream>>>(
      x_bf, Wqkv_bf, bqkv, nullptr, qkv_bf, B_ * T_, 3 * C_, C_);

  // attention + gate -> y_bf
  attn_kernel<<<B_ * H_, 128, 0, stream>>>(qkv_bf, mem_bf, gate, y_bf);

  // out = (y*gate) @ Wproj + bproj, scaled by loss_mask  -> f32
  gemm_bf16_kernel<0><<<dim3(C_ / 64, (B_ * T_) / 128), 256, 0, stream>>>(
      y_bf, Wproj_bf, bproj, loss_mask, out, B_ * T_, C_, C_);
}
